// NodeEdgeEarlyInteraction_43465069035814
// MI455X (gfx1250) — matrix operand order repaired and checked by reading, not compiled, not run on hardware
//
#include <hip/hip_runtime.h>

// ---- problem constants (static for this reference setup) ----
#define GN     256      // graphs
#define BPH    128      // graph pairs
#define NNODE  7680     // total nodes  (GN*NPG)
#define NEDGE  30720    // total edges  (GN*EPG)
#define DN_    64
#define DM_    128
#define RAW_   32
#define SKD_   32
#define NPG_   30
#define EPG_   120
#define NSD    256      // node_store width  DN*(T+1)
#define ESD    512      // edge_store width  DM*(T+1)
#define TEMP_  0.1f
#define SITERS_ 10

typedef __attribute__((ext_vector_type(16))) _Float16 v16h;
typedef __attribute__((ext_vector_type(8)))  _Float16 v8h;
typedef __attribute__((ext_vector_type(8)))  float    v8f;

// ---- CDNA5 async global->LDS staging (ASYNCcnt-tracked, no VGPR return) ----
__device__ __forceinline__ void async_f32_to_lds(const float* gsrc, void* lds_dst) {
  unsigned loff = (unsigned)(unsigned long long)lds_dst;   // flat addr truncates to LDS offset
  asm volatile("global_load_async_to_lds_b32 %0, %1, off"
               :: "v"(loff), "v"(gsrc) : "memory");
}
__device__ __forceinline__ void async_wait0() {
  asm volatile("s_wait_asynccnt 0x0" ::: "memory");
}

// =====================================================================
// Generic WMMA GEMM:  C[M,Nc] = act(Af16[M,K] @ W[K,Nc] + bias)
// W supplied transposed+f16: Wt[Nc,K].  One 16x32 tile per wave
// (one A fragment reused by two WMMAs per K-step).
// Output supports row stride ldc and column offset c0.
// =====================================================================
__global__ __launch_bounds__(256) void k_gemm(
    const _Float16* __restrict__ A, const _Float16* __restrict__ Wt,
    const float* __restrict__ bias, float* __restrict__ C,
    int ldc, int c0, int M, int K, int Nc, int relu)
{
  const int lane  = threadIdx.x & 31;
  const int wave  = threadIdx.x >> 5;
  const int tilesN2 = Nc >> 5;                 // 32-wide N tiles
  const int tile  = blockIdx.x * 8 + wave;
  const int totTiles = (M >> 4) * tilesN2;
  if (tile >= totTiles) return;
  const int tm = tile / tilesN2, tn = tile % tilesN2;
  const int hn = lane & 15, cc = lane >> 4;

  const _Float16* arow  = A  + (size_t)(tm * 16 + hn) * K;
  const _Float16* wrow0 = Wt + (size_t)(tn * 32 + hn) * K + cc * 16;
  const _Float16* wrow1 = wrow0 + (size_t)16 * K;
  __builtin_prefetch(arow, 0, 1);

  v8f acc0 = {}, acc1 = {};
  union { v16h v; v8h h[2]; } a;
  for (int k0 = 0; k0 < K; k0 += 32) {
    // 16-bit A fragment: lane = row M, 16 consecutive K values (cc selects K half)
    a.h[0] = *(const v8h*)(arow + k0 + cc * 16);
    a.h[1] = *(const v8h*)(arow + k0 + cc * 16 + 8);
    // B fragments: lane = column, 16 consecutive K values (cc selects K half)
    v16h b0 = *(const v16h*)(wrow0 + k0);
    v16h b1 = *(const v16h*)(wrow1 + k0);
    acc0 = __builtin_amdgcn_wmma_f32_16x16x32_f16(
        false, a.v, false, b0, (short)0, acc0, false, false);
    acc1 = __builtin_amdgcn_wmma_f32_16x16x32_f16(
        false, a.v, false, b1, (short)0, acc1, false, false);
  }
  const int colBase = tn * 32 + hn;
  const float bv0 = bias[colBase];
  const float bv1 = bias[colBase + 16];
  float* cptr = C + (size_t)(tm * 16 + cc * 8) * ldc + c0 + colBase;
#pragma unroll
  for (int r = 0; r < 8; ++r) {
    float v0 = acc0[r] + bv0;
    float v1 = acc1[r] + bv1;
    if (relu) { v0 = v0 > 0.f ? v0 : 0.f; v1 = v1 > 0.f ? v1 : 0.f; }
    cptr[(size_t)r * ldc]      = v0;
    cptr[(size_t)r * ldc + 16] = v1;
  }
}

// ---- f32 -> f16 convert ----
__global__ void k_cvt(const float* __restrict__ in, _Float16* __restrict__ out, long long n) {
  long long i = (long long)blockIdx.x * blockDim.x + threadIdx.x;
  if (i < n) out[i] = (_Float16)in[i];
}

// ---- pack concat [a | b_slice] -> f16 ----
__global__ void k_pack2(const float* __restrict__ a, int ka,
                        const float* __restrict__ b, int ldb, int b0, int kb,
                        _Float16* __restrict__ out, int M) {
  const int K = ka + kb;
  long long idx = (long long)blockIdx.x * blockDim.x + threadIdx.x;
  if (idx >= (long long)M * K) return;
  int i = (int)(idx / K), k = (int)(idx % K);
  float v = (k < ka) ? a[(long long)i * ka + k]
                     : b[(long long)i * ldb + b0 + (k - ka)];
  out[idx] = (_Float16)v;
}

// ---- pack message input [ncf[from] | ncf[to] | ecf] -> f16 (E x 192) ----
__global__ void k_pack_msg(const float* __restrict__ ncf, const float* __restrict__ ecf,
                           const int* __restrict__ from, const int* __restrict__ to,
                           _Float16* __restrict__ out) {
  long long idx = (long long)blockIdx.x * blockDim.x + threadIdx.x;
  if (idx >= (long long)NEDGE * 192) return;
  int e = (int)(idx / 192), k = (int)(idx % 192);
  float v;
  if (k < 64)       v = ncf[(long long)from[e] * 64 + k];
  else if (k < 128) v = ncf[(long long)to[e]   * 64 + (k - 64)];
  else              v = ecf[(long long)e * 64 + (k - 128)];
  out[idx] = (_Float16)v;
}

// ---- pack padded Sinkhorn features: rows 0..4095 = q, 4096..8191 = c (8192 x 64 f16)
__global__ void k_pack_sk(const float* __restrict__ un, _Float16* __restrict__ out) {
  long long idx = (long long)blockIdx.x * blockDim.x + threadIdx.x;
  if (idx >= 8192LL * 64) return;
  int r = (int)(idx >> 6), d = (int)(idx & 63);
  int which = r >> 12;          // 0 = query graphs (even), 1 = corpus (odd)
  int b = (r & 4095) >> 5;
  int i = r & 31;
  float v = 0.f;
  if (i < NPG_) v = un[((long long)(2 * b + which) * NPG_ + i) * NSD + 192 + d];
  out[idx] = (_Float16)v;
}

// ---- copy dense block into strided slice ----
__global__ void k_copy_slice(const float* __restrict__ src, int lds_,
                             float* __restrict__ dst, int ldd, int d0, int M, int W) {
  long long idx = (long long)blockIdx.x * blockDim.x + threadIdx.x;
  if (idx >= (long long)M * W) return;
  int i = (int)(idx / W), w = (int)(idx % W);
  dst[(long long)i * ldd + d0 + w] = src[(long long)i * lds_ + w];
}

// ---- segment sum (scatter-add messages into nodes) ----
__global__ void k_segsum(const float* __restrict__ msg, const int* __restrict__ to,
                         float* __restrict__ agg) {
  long long idx = (long long)blockIdx.x * blockDim.x + threadIdx.x;
  if (idx >= (long long)NEDGE * DM_) return;
  int e = (int)(idx >> 7), f = (int)(idx & 127);
  unsafeAtomicAdd(&agg[(long long)to[e] * DM_ + f], msg[idx]);
}

// ---- GRU elementwise ----
__global__ void k_gru(const float* __restrict__ gi, const float* __restrict__ gh,
                      const float* __restrict__ h, float* __restrict__ out) {
  long long idx = (long long)blockIdx.x * blockDim.x + threadIdx.x;
  if (idx >= (long long)NNODE * DN_) return;
  int i = (int)(idx >> 6), d = (int)(idx & 63);
  const float* gip = gi + (long long)i * 192;
  const float* ghp = gh + (long long)i * 192;
  float r = 1.f / (1.f + __expf(-(gip[d] + ghp[d])));
  float z = 1.f / (1.f + __expf(-(gip[64 + d] + ghp[64 + d])));
  float n = tanhf(gip[128 + d] + r * ghp[128 + d]);
  out[idx] = (1.f - z) * n + z * h[idx];
}

// ---- masked node match scores: scores[b,i,j] = <tqf[b,i], tcf[b,j]> ----
__global__ void k_nscore(const float* __restrict__ tqc, float* __restrict__ scores) {
  int b = blockIdx.x;
  for (int t = threadIdx.x; t < 1024; t += blockDim.x) {
    int i = t >> 5, j = t & 31;
    float v = 0.f;
    if (i < NPG_ && j < NPG_) {
      const float* q = tqc + ((long long)b * 32 + i) * 32;
      const float* c = tqc + (4096LL + (long long)b * 32 + j) * 32;
#pragma unroll
      for (int d = 0; d < 32; ++d) v += q[d] * c[d];
    }
    scores[(long long)b * 1024 + t] = v;
  }
}

// ---- Sinkhorn 32x32, one wave per pair ----
__global__ __launch_bounds__(32) void k_sinkhorn32(const float* __restrict__ sc,
                                                   float* __restrict__ plan) {
  __shared__ float la[32][33];
  int b = blockIdx.x, t = threadIdx.x;
  for (int j = 0; j < 32; ++j) la[t][j] = sc[(long long)b * 1024 + t * 32 + j] * (1.f / TEMP_);
  __syncthreads();
  for (int it = 0; it < SITERS_; ++it) {
    float m = -1e30f;
    for (int j = 0; j < 32; ++j) m = fmaxf(m, la[t][j]);
    float s = 0.f;
    for (int j = 0; j < 32; ++j) s += __expf(la[t][j] - m);
    float l = m + __logf(s);
    for (int j = 0; j < 32; ++j) la[t][j] -= l;
    __syncthreads();
    m = -1e30f;
    for (int i = 0; i < 32; ++i) m = fmaxf(m, la[i][t]);
    s = 0.f;
    for (int i = 0; i < 32; ++i) s += __expf(la[i][t] - m);
    l = m + __logf(s);
    for (int i = 0; i < 32; ++i) la[i][t] -= l;
    __syncthreads();
  }
  for (int j = 0; j < 32; ++j) plan[(long long)b * 1024 + t * 32 + j] = __expf(la[t][j]);
}

// ---- edge match scores from node plan (async-staged plan tile) ----
__global__ __launch_bounds__(256) void k_escore(const float* __restrict__ plan,
    const int* __restrict__ from, const int* __restrict__ to, float* __restrict__ esc) {
  __shared__ float sp[1024];
  __shared__ int fq[EPG_], tq[EPG_], fc[EPG_], tc[EPG_];
  int b = blockIdx.x;
  for (int t = threadIdx.x; t < 1024; t += 256)
    async_f32_to_lds(&plan[(long long)b * 1024 + t], &sp[t]);
  for (int t = threadIdx.x; t < EPG_; t += 256) {
    int e0 = (2 * b) * EPG_ + t, e1 = (2 * b + 1) * EPG_ + t;
    fq[t] = from[e0] - (2 * b) * NPG_;     tq[t] = to[e0] - (2 * b) * NPG_;
    fc[t] = from[e1] - (2 * b + 1) * NPG_; tc[t] = to[e1] - (2 * b + 1) * NPG_;
  }
  async_wait0();
  __syncthreads();
  for (int t = threadIdx.x; t < 128 * 128; t += 256) {
    int i = t >> 7, j = t & 127;
    float v = 0.f;
    if (i < EPG_ && j < EPG_) {
      v = sp[fq[i] * 32 + fc[j]] * sp[tq[i] * 32 + tc[j]]
        + sp[fq[i] * 32 + tc[j]] * sp[tq[i] * 32 + fc[j]];
    }
    esc[(long long)b * 16384 + t] = v;
  }
}

// ---- Sinkhorn 128x128, padded LDS (stride 129) ----
__global__ __launch_bounds__(128) void k_sinkhorn128(const float* __restrict__ sc,
                                                     float* __restrict__ plan) {
  extern __shared__ float la[];   // 128*129 floats
  int b = blockIdx.x, t = threadIdx.x;
  for (int j = 0; j < 128; ++j) la[t * 129 + j] = sc[(long long)b * 16384 + t * 128 + j] * (1.f / TEMP_);
  __syncthreads();
  for (int it = 0; it < SITERS_; ++it) {
    float m = -1e30f;
    for (int j = 0; j < 128; ++j) m = fmaxf(m, la[t * 129 + j]);
    float s = 0.f;
    for (int j = 0; j < 128; ++j) s += __expf(la[t * 129 + j] - m);
    float l = m + __logf(s);
    for (int j = 0; j < 128; ++j) la[t * 129 + j] -= l;
    __syncthreads();
    m = -1e30f;
    for (int i = 0; i < 128; ++i) m = fmaxf(m, la[i * 129 + t]);
    s = 0.f;
    for (int i = 0; i < 128; ++i) s += __expf(la[i * 129 + t] - m);
    l = m + __logf(s);
    for (int i = 0; i < 128; ++i) la[i * 129 + t] -= l;
    __syncthreads();
  }
  for (int j = 0; j < 128; ++j) plan[(long long)b * 16384 + t * 128 + j] = __expf(la[t * 129 + j]);
}

// ---- node interact + scatter: node_store[:,64:] = inter[pad_n,64:] ----
__global__ void k_ninter(const float* __restrict__ plan, const float* __restrict__ un,
                         float* __restrict__ ns) {
  long long idx = (long long)blockIdx.x * blockDim.x + threadIdx.x;
  if (idx >= (long long)NNODE * 192) return;
  int j = (int)(idx / 192), f = (int)(idx % 192);
  int g = j / NPG_, l = j % NPG_, b = g >> 1;
  const float* P = plan + (long long)b * 1024;
  float acc = 0.f;
  if (!(g & 1)) {   // query graph: qn = plan @ cs
    const float* base = un + ((long long)(2 * b + 1) * NPG_) * NSD + 64 + f;
    for (int jj = 0; jj < NPG_; ++jj) acc += P[l * 32 + jj] * base[(long long)jj * NSD];
  } else {          // corpus graph: cn = plan^T @ qs
    const float* base = un + ((long long)(2 * b) * NPG_) * NSD + 64 + f;
    for (int ii = 0; ii < NPG_; ++ii) acc += P[ii * 32 + l] * base[(long long)ii * NSD];
  }
  ns[(long long)j * NSD + 64 + f] = acc;
}

// ---- edge interact + scatter: edge_store[:,128:] = einter[pad_e,128:] ----
__global__ __launch_bounds__(384) void k_einter(const float* __restrict__ eplan,
    const float* __restrict__ ue, float* __restrict__ es) {
  __shared__ float prow[128];
  int eid = blockIdx.x;                 // global edge index
  int g = eid / EPG_, l = eid % EPG_, b = g >> 1;
  const float* P = eplan + (long long)b * 16384;
  if (threadIdx.x < 128) {
    const float* src = (g & 1) ? &P[threadIdx.x * 128 + l] : &P[l * 128 + threadIdx.x];
    async_f32_to_lds(src, &prow[threadIdx.x]);
    async_wait0();
  }
  __syncthreads();
  int f = threadIdx.x;                  // 0..383 -> cols 128..511
  const float* base = ue + ((long long)((g & 1) ? (2 * b) * EPG_ : (2 * b + 1) * EPG_)) * ESD + 128 + f;
  float acc = 0.f;
  for (int jj = 0; jj < EPG_; ++jj) acc += prow[jj] * base[(long long)jj * ESD];
  es[(long long)eid * ESD + 128 + f] = acc;
}

// ---- final score ----
__global__ __launch_bounds__(64) void k_score(const float* __restrict__ plan,
    const float* __restrict__ un, float* __restrict__ out) {
  __shared__ float sp[1024];
  __shared__ float fcs[32][65];
  __shared__ float red[64];
  int b = blockIdx.x, t = threadIdx.x;  // t = feature dim d (0..63)
  for (int k = t; k < 1024; k += 64) sp[k] = plan[(long long)b * 1024 + k];
  for (int j = 0; j < 32; ++j)
    fcs[j][t] = (j < NPG_) ? un[((long long)(2 * b + 1) * NPG_ + j) * NSD + 192 + t] : 0.f;
  __syncthreads();
  float acc = 0.f;
  for (int i = 0; i < 32; ++i) {
    float fqv = (i < NPG_) ? un[((long long)(2 * b) * NPG_ + i) * NSD + 192 + t] : 0.f;
    float m = 0.f;
    for (int j = 0; j < 32; ++j) m += sp[i * 32 + j] * fcs[j][t];
    float d = fqv - m;
    acc += d > 0.f ? d : 0.f;
  }
  red[t] = acc; __syncthreads();
  for (int s = 32; s > 0; s >>= 1) { if (t < s) red[t] += red[t + s]; __syncthreads(); }
  if (t == 0) out[b] = -red[0];
}

// ---- weight transpose + f16 convert: Wt[n,k] = (f16)W[k,n] ----
__global__ void k_wt(const float* __restrict__ W, _Float16* __restrict__ Wt, int K, int Nc) {
  int idx = blockIdx.x * blockDim.x + threadIdx.x;
  if (idx >= K * Nc) return;
  int n = idx / K, k = idx % K;
  Wt[idx] = (_Float16)W[(long long)k * Nc + n];
}

// =====================================================================
extern "C" void kernel_launch(void* const* d_in, const int* in_sizes, int n_in,
                              void* d_out, int out_size, void* d_ws, size_t ws_size,
                              hipStream_t stream)
{
  (void)in_sizes; (void)n_in; (void)out_size; (void)ws_size;
  const float* node_features = (const float*)d_in[0];
  const float* edge_features = (const float*)d_in[1];
  const float* enc_nw = (const float*)d_in[2];  const float* enc_nb = (const float*)d_in[3];
  const float* enc_ew = (const float*)d_in[4];  const float* enc_eb = (const float*)d_in[5];
  const float* ni_w1  = (const float*)d_in[6];  const float* ni_b1  = (const float*)d_in[7];
  const float* ni_w2  = (const float*)d_in[8];  const float* ni_b2  = (const float*)d_in[9];
  const float* ei_w1  = (const float*)d_in[10]; const float* ei_b1  = (const float*)d_in[11];
  const float* ei_w2  = (const float*)d_in[12]; const float* ei_b2  = (const float*)d_in[13];
  const float* msg_w1 = (const float*)d_in[14]; const float* msg_b1 = (const float*)d_in[15];
  const float* msg_w2 = (const float*)d_in[16]; const float* msg_b2 = (const float*)d_in[17];
  const float* gru_wih = (const float*)d_in[18]; const float* gru_bih = (const float*)d_in[19];
  const float* gru_whh = (const float*)d_in[20]; const float* gru_bhh = (const float*)d_in[21];
  const float* sk_w1  = (const float*)d_in[22]; const float* sk_b1  = (const float*)d_in[23];
  const float* sk_w2  = (const float*)d_in[24]; const float* sk_b2  = (const float*)d_in[25];
  const int* from_idx = (const int*)d_in[26];
  const int* to_idx   = (const int*)d_in[27];

  // ---- workspace carve ----
  char* cur = (char*)d_ws;
  auto alloc = [&](size_t bytes) -> char* {
    char* p = cur; cur += (bytes + 255) & ~(size_t)255; return p;
  };
  _Float16* wtEncN = (_Float16*)alloc(64 * 32 * 2);
  _Float16* wtEncE = (_Float16*)alloc(64 * 32 * 2);
  _Float16* wtNi1  = (_Float16*)alloc(128 * 128 * 2);
  _Float16* wtNi2  = (_Float16*)alloc(64 * 128 * 2);
  _Float16* wtEi1  = (_Float16*)alloc(192 * 192 * 2);
  _Float16* wtEi2  = (_Float16*)alloc(64 * 192 * 2);
  _Float16* wtMsg1 = (_Float16*)alloc(128 * 192 * 2);
  _Float16* wtMsg2 = (_Float16*)alloc(128 * 128 * 2);
  _Float16* wtWih  = (_Float16*)alloc(192 * 128 * 2);
  _Float16* wtWhh  = (_Float16*)alloc(192 * 64 * 2);
  _Float16* wtSk1  = (_Float16*)alloc(32 * 64 * 2);
  _Float16* wtSk2  = (_Float16*)alloc(32 * 32 * 2);
  float* enc_n   = (float*)alloc((size_t)NNODE * 64 * 4);
  float* enc_e   = (float*)alloc((size_t)NEDGE * 64 * 4);
  float* nstore  = (float*)alloc((size_t)NNODE * NSD * 4);
  float* unode   = (float*)alloc((size_t)NNODE * NSD * 4);
  float* estore  = (float*)alloc((size_t)NEDGE * ESD * 4);
  float* uedge   = (float*)alloc((size_t)NEDGE * ESD * 4);
  float* ncf     = (float*)alloc((size_t)NNODE * 64 * 4);
  float* nfe     = (float*)alloc((size_t)NNODE * 64 * 4);
  float* ecf     = (float*)alloc((size_t)NEDGE * 64 * 4);
  float* msgb    = (float*)alloc((size_t)NEDGE * 128 * 4);
  float* agg     = (float*)alloc((size_t)NNODE * 128 * 4);
  float* gi      = (float*)alloc((size_t)NNODE * 192 * 4);
  float* gh      = (float*)alloc((size_t)NNODE * 192 * 4);
  float* tmpN    = (float*)alloc((size_t)NNODE * 128 * 4);
  float* tmpE    = (float*)alloc((size_t)NEDGE * 192 * 4);
  _Float16* packA = (_Float16*)alloc((size_t)NEDGE * 192 * 2);
  float* tqc     = (float*)alloc((size_t)8192 * 32 * 4);
  float* scores  = (float*)alloc((size_t)BPH * 1024 * 4);
  float* plan    = (float*)alloc((size_t)BPH * 1024 * 4);
  float* escr    = (float*)alloc((size_t)BPH * 16384 * 4);
  float* eplan   = (float*)alloc((size_t)BPH * 16384 * 4);

  auto gemm = [&](const _Float16* A, const _Float16* Wt, const float* bias, float* C,
                  int ldc, int c0, int M, int K, int Nc, int relu) {
    int tiles = (M / 16) * (Nc / 32);
    k_gemm<<<(tiles + 7) / 8, 256, 0, stream>>>(A, Wt, bias, C, ldc, c0, M, K, Nc, relu);
  };
  auto cvt = [&](const float* in, _Float16* out, long long n) {
    k_cvt<<<(unsigned)((n + 255) / 256), 256, 0, stream>>>(in, out, n);
  };
  auto cvtW = [&](const float* W, _Float16* Wt, int K, int Nc) {
    int n = K * Nc; k_wt<<<(n + 255) / 256, 256, 0, stream>>>(W, Wt, K, Nc);
  };

  // ---- convert all weights once (transposed f16) ----
  cvtW(enc_nw, wtEncN, 32, 64);   cvtW(enc_ew, wtEncE, 32, 64);
  cvtW(ni_w1, wtNi1, 128, 128);   cvtW(ni_w2, wtNi2, 128, 64);
  cvtW(ei_w1, wtEi1, 192, 192);   cvtW(ei_w2, wtEi2, 192, 64);
  cvtW(msg_w1, wtMsg1, 192, 128); cvtW(msg_w2, wtMsg2, 128, 128);
  cvtW(gru_wih, wtWih, 128, 192); cvtW(gru_whh, wtWhh, 64, 192);
  cvtW(sk_w1, wtSk1, 64, 32);     cvtW(sk_w2, wtSk2, 32, 32);

  // ---- zero stores (cols [0,64)/[0,128) stay zero forever) ----
  hipMemsetAsync(nstore, 0, (size_t)NNODE * NSD * 4, stream);
  hipMemsetAsync(unode,  0, (size_t)NNODE * NSD * 4, stream);
  hipMemsetAsync(estore, 0, (size_t)NEDGE * ESD * 4, stream);
  hipMemsetAsync(uedge,  0, (size_t)NEDGE * ESD * 4, stream);

  // ---- encoders ----
  cvt(node_features, packA, (long long)NNODE * RAW_);
  gemm(packA, wtEncN, enc_nb, enc_n, 64, 0, NNODE, 32, 64, 0);
  cvt(edge_features, packA, (long long)NEDGE * RAW_);
  gemm(packA, wtEncE, enc_eb, enc_e, 64, 0, NEDGE, 32, 64, 0);

  const unsigned gN64  = (unsigned)(((long long)NNODE * 64 + 255) / 256);
  const unsigned gN128 = (unsigned)(((long long)NNODE * 128 + 255) / 256);
  const unsigned gN192 = (unsigned)(((long long)NNODE * 192 + 255) / 256);
  const unsigned gE128 = (unsigned)(((long long)NEDGE * 128 + 255) / 256);
  const unsigned gE192 = (unsigned)(((long long)NEDGE * 192 + 255) / 256);

  for (int tu = 0; tu < 2; ++tu) {
    // ncf = node_inter([enc_n, node_store[:, 0:64]])
    k_pack2<<<gN128, 256, 0, stream>>>(enc_n, 64, nstore, NSD, 0, 64, packA, NNODE);
    gemm(packA, wtNi1, ni_b1, tmpN, 128, 0, NNODE, 128, 128, 1);
    cvt(tmpN, packA, (long long)NNODE * 128);
    gemm(packA, wtNi2, ni_b2, ncf, 64, 0, NNODE, 128, 64, 0);

    for (int step = 1; step <= 3; ++step) {
      const int ei = 128 * step, ni = 64 * step;
      // ecf = edge_inter([enc_e, edge_store[:, ei-128:ei]])
      k_pack2<<<gE192, 256, 0, stream>>>(enc_e, 64, estore, ESD, ei - 128, 128, packA, NEDGE);
      gemm(packA, wtEi1, ei_b1, tmpE, 192, 0, NEDGE, 192, 192, 1);
      cvt(tmpE, packA, (long long)NEDGE * 192);
      gemm(packA, wtEi2, ei_b2, ecf, 64, 0, NEDGE, 192, 64, 0);
      // msg = message(ncf, ecf)
      k_pack_msg<<<gE192, 256, 0, stream>>>(ncf, ecf, from_idx, to_idx, packA);
      gemm(packA, wtMsg1, msg_b1, tmpE, 128, 0, NEDGE, 192, 128, 1);
      cvt(tmpE, packA, (long long)NEDGE * 128);
      gemm(packA, wtMsg2, msg_b2, msgb, 128, 0, NEDGE, 128, 128, 0);
      // agg = segment_sum(msg, to_idx)
      hipMemsetAsync(agg, 0, (size_t)NNODE * 128 * 4, stream);
      k_segsum<<<gE128, 256, 0, stream>>>(msgb, to_idx, agg);
      // nfe = GRU(agg, ncf)
      cvt(agg, packA, (long long)NNODE * 128);
      gemm(packA, wtWih, gru_bih, gi, 192, 0, NNODE, 128, 192, 0);
      cvt(ncf, packA, (long long)NNODE * 64);
      gemm(packA, wtWhh, gru_bhh, gh, 192, 0, NNODE, 64, 192, 0);
      k_gru<<<gN64, 256, 0, stream>>>(gi, gh, ncf, nfe);
      // upd_node[:, ni:ni+64] = nfe
      k_copy_slice<<<gN64, 256, 0, stream>>>(nfe, 64, unode, NSD, ni, NNODE, 64);
      // ncf = node_inter([nfe, node_store[:, ni:ni+64]])
      k_pack2<<<gN128, 256, 0, stream>>>(nfe, 64, nstore, NSD, ni, 64, packA, NNODE);
      gemm(packA, wtNi1, ni_b1, tmpN, 128, 0, NNODE, 128, 128, 1);
      cvt(tmpN, packA, (long long)NNODE * 128);
      gemm(packA, wtNi2, ni_b2, ncf, 64, 0, NNODE, 128, 64, 0);
      // upd_edge[:, ei:ei+128] = message(ncf, ecf) (write directly into slice)
      k_pack_msg<<<gE192, 256, 0, stream>>>(ncf, ecf, from_idx, to_idx, packA);
      gemm(packA, wtMsg1, msg_b1, tmpE, 128, 0, NEDGE, 192, 128, 1);
      cvt(tmpE, packA, (long long)NEDGE * 128);
      gemm(packA, wtMsg2, msg_b2, uedge, ESD, ei, NEDGE, 128, 128, 0);
    }

    // ---- Sinkhorn matching head ----
    k_pack_sk<<<(unsigned)((8192LL * 64 + 255) / 256), 256, 0, stream>>>(unode, packA);
    gemm(packA, wtSk1, sk_b1, tmpN, 32, 0, 8192, 64, 32, 1);
    cvt(tmpN, packA, 8192LL * 32);
    gemm(packA, wtSk2, sk_b2, tqc, 32, 0, 8192, 32, 32, 0);
    k_nscore<<<BPH, 256, 0, stream>>>(tqc, scores);
    k_sinkhorn32<<<BPH, 32, 0, stream>>>(scores, plan);
    // node_store = upd_node with cols 64.. replaced by interaction
    k_ninter<<<gN192, 256, 0, stream>>>(plan, unode, nstore);
    // edge plan + edge interaction
    k_escore<<<BPH, 256, 0, stream>>>(plan, from_idx, to_idx, escr);
    k_sinkhorn128<<<BPH, 128, 128 * 129 * 4, stream>>>(escr, eplan);
    k_einter<<<NEDGE, 384, 0, stream>>>(eplan, uedge, estore);
  }

  // ---- final score (uses last plan and last upd_node features) ----
  k_score<<<BPH, 64, 0, stream>>>(plan, unode, (float*)d_out);
}
